// CausalSelfAttention_87608742904617
// MI455X (gfx1250) — matrix operand order repaired and checked by reading, not compiled, not run on hardware
//
#include <hip/hip_runtime.h>
#include <hip/hip_bf16.h>
#include <math.h>

// Problem constants (B,S,D,H from reference)
#define B_  2
#define S_  2048
#define D_  2048
#define H_  16
#define DK_ 128

// LDS tile row pitch (floats): 32 data + 4 pad to spread banks
#define TPITCH 36

typedef __attribute__((ext_vector_type(16))) __bf16 v16bf;
typedef __attribute__((ext_vector_type(8)))  __bf16 v8bf;
typedef __attribute__((ext_vector_type(8)))  float  v8f;
typedef int v4i __attribute__((vector_size(16)));   // 128-bit chunk type for async-LDS builtin

#if defined(__has_builtin)
#if __has_builtin(__builtin_amdgcn_global_load_async_to_lds_b128)
#define ASYNC_LDS 1
#endif
#endif

#if defined(__has_builtin) && __has_builtin(__builtin_amdgcn_s_wait_asynccnt)
#define WAIT_ASYNC_LE(n) __builtin_amdgcn_s_wait_asynccnt(n)
#else
#define WAIT_ASYNC_LE(n) asm volatile("s_wait_asynccnt " #n ::: "memory")
#endif

__device__ __forceinline__ int lane_id() { return threadIdx.x & 31; }

__device__ __forceinline__ v8f vzero8() {
  v8f z;
#pragma unroll
  for (int r = 0; r < 8; ++r) z[r] = 0.0f;
  return z;
}

// ---------------------------------------------------------------------------
// WMMA bf16 fragment loaders (wave32 layouts per CDNA5 ISA 7.12.2)
// A: 16x32 (MxK). lane L: m=L&15, kb=(L<16?0:8). VGPR0..3 hold K=kb..kb+7,
//    VGPR4..7 hold K=kb+16..kb+23.  -> two contiguous 8-elem runs.
// B: 32x16 (KxN). lane L: n=L&15, kb=(L<16?0:8). elems 0..7 hold K=kb..kb+7,
//    elems 8..15 hold K=kb+16..kb+23 from "row n" of an [N][K] row-major source.
// ---------------------------------------------------------------------------
__device__ __forceinline__ v16bf load_a_bf16(const __bf16* A, int lda) {
  const int l = lane_id();
  const int m = l & 15;
  const int kb = (l < 16) ? 0 : 8;
  const __bf16* p = A + (size_t)m * lda + kb;
  v8bf lo = *(const v8bf*)p;
  v8bf hi = *(const v8bf*)(p + 16);
  v16bf r;
#pragma unroll
  for (int i = 0; i < 8; ++i) { r[i] = lo[i]; r[i + 8] = hi[i]; }
  return r;
}

__device__ __forceinline__ v16bf load_a_f32(const float* A, int lda) {
  const int l = lane_id();
  const int m = l & 15;
  const int kb = (l < 16) ? 0 : 8;
  const float* p = A + (size_t)m * lda + kb;
  v16bf r;
#pragma unroll
  for (int i = 0; i < 8; ++i) { r[i] = (__bf16)p[i]; r[i + 8] = (__bf16)p[16 + i]; }
  return r;
}

__device__ __forceinline__ v16bf load_b_bf16(const __bf16* Bm, int ldb) {
  const int l = lane_id();
  const int n = l & 15;
  const int kb = (l < 16) ? 0 : 8;
  const __bf16* p = Bm + (size_t)n * ldb + kb;
  v8bf x0 = *(const v8bf*)p;
  v8bf x1 = *(const v8bf*)(p + 16);
  v16bf r;
#pragma unroll
  for (int i = 0; i < 8; ++i) { r[i] = x0[i]; r[i + 8] = x1[i]; }
  return r;
}

// Fragment loaders reading a staged f32 tile in LDS (row pitch TPITCH floats),
// converting to bf16 in-register (ds_load_b128 + v_cvt_pk_bf16_f32).
__device__ __forceinline__ v16bf lds_frag_a(const float* p0) {
  const int l = lane_id();
  const int m = l & 15;
  const int kb = (l < 16) ? 0 : 8;
  const float* p = p0 + m * TPITCH + kb;
  v16bf r;
#pragma unroll
  for (int i = 0; i < 8; ++i) { r[i] = (__bf16)p[i]; r[i + 8] = (__bf16)p[16 + i]; }
  return r;
}

__device__ __forceinline__ v16bf lds_frag_b(const float* p0) {
  const int l = lane_id();
  const int n = l & 15;
  const int kb = (l < 16) ? 0 : 8;
  const float* p = p0 + n * TPITCH + kb;
  v16bf r;
#pragma unroll
  for (int i = 0; i < 8; ++i) { r[i] = (__bf16)p[i]; r[i + 8] = (__bf16)p[16 + i]; }
  return r;
}

// ---------------------------------------------------------------------------
// Stage one 128-row x 32-float f32 tile from global into LDS (pitch TPITCH).
// 1024 x 16B chunks, 256 threads x 4 chunks. Async DMA when available
// (GLOBAL_LOAD_ASYNC_TO_LDS_B128, ASYNCcnt), else load+ds_store fallback.
// ---------------------------------------------------------------------------
__device__ __forceinline__ void stage_tile(const float* __restrict__ g, int ldg,
                                           float* l, int t)
{
#pragma unroll
  for (int i = 0; i < 4; ++i) {
    const int c = i * 256 + t;
    const int row = c >> 3;
    const int seg = c & 7;
    const float* gp = g + (size_t)row * ldg + seg * 4;
    float* lp = l + row * TPITCH + seg * 4;
#ifdef ASYNC_LDS
    __builtin_amdgcn_global_load_async_to_lds_b128((v4i*)gp, (v4i*)lp, 0, 0);
#else
    *(float4*)lp = *(const float4*)gp;
#endif
  }
}

// ---------------------------------------------------------------------------
// Projection GEMM: C[m][n] = sum_k A[m][k] * W[n][k] + bias[n]
// M = B*S = 4096, N = K = D = 2048. Block 256thr = 8 waves (4 M x 2 N),
// block tile 128x128, wave tile 32x64, K step 32.
// Double-buffered async LDS staging of W (and A when A is f32).
// mode 0: write bf16 row-major [M][N]    (Q / K projections)
// mode 1: write bf16 V-transposed  Vt[b][h][dk][s]   (V projection)
// mode 2: write f32 row-major [M][N]     (output projection, A is bf16)
// ---------------------------------------------------------------------------
__global__ void __launch_bounds__(256)
proj_gemm_kernel(const float* __restrict__ A32, const __bf16* __restrict__ A16,
                 const float* __restrict__ W, const float* __restrict__ bias,
                 __bf16* __restrict__ outB, float* __restrict__ outF, int mode)
{
  __shared__ float Asm[2][128 * TPITCH];   // 36 KB
  __shared__ float Bsm[2][128 * TPITCH];   // 36 KB  (72 KB of 320 KB/WGP)

  const int K = D_;
  const int t = threadIdx.x;
  const int wave = t >> 5;
  const int mBlk = blockIdx.y * 128;
  const int nBlk = blockIdx.x * 128;
  const int m0w = (wave >> 1) * 32;
  const int n0w = (wave & 1) * 64;
  const bool useA16 = (A16 != nullptr);

  v8f c[2][4];
#pragma unroll
  for (int i = 0; i < 2; ++i)
#pragma unroll
    for (int j = 0; j < 4; ++j) c[i][j] = vzero8();

  // Prologue: stage k0 = 0 into buffer 0.
  stage_tile(W + (size_t)nBlk * K, K, &Bsm[0][0], t);
  if (!useA16) stage_tile(A32 + (size_t)mBlk * K, K, &Asm[0][0], t);

  const int NT = K / 32;   // 64
  int buf = 0;
  for (int kt = 0; kt < NT; ++kt) {
    const int cur = buf, nxt = buf ^ 1;
    const bool hn = (kt + 1 < NT);
    if (hn) {
      const int kn = (kt + 1) * 32;
      stage_tile(W + (size_t)nBlk * K + kn, K, &Bsm[nxt][0], t);
      if (!useA16) stage_tile(A32 + (size_t)mBlk * K + kn, K, &Asm[nxt][0], t);
      // ASYNCcnt retires in issue order: allow next stage's chunks in flight,
      // guarantee current stage's chunks have landed.
      if (useA16) { WAIT_ASYNC_LE(4); } else { WAIT_ASYNC_LE(8); }
    } else {
      WAIT_ASYNC_LE(0);
    }
    __syncthreads();

    v16bf af[2], bfr[4];
    if (useA16) {
      const int k0 = kt * 32;
      af[0] = load_a_bf16(A16 + (size_t)(mBlk + m0w) * K + k0, K);
      af[1] = load_a_bf16(A16 + (size_t)(mBlk + m0w + 16) * K + k0, K);
    } else {
      af[0] = lds_frag_a(&Asm[cur][(size_t)m0w * TPITCH]);
      af[1] = lds_frag_a(&Asm[cur][(size_t)(m0w + 16) * TPITCH]);
    }
#pragma unroll
    for (int j = 0; j < 4; ++j)
      bfr[j] = lds_frag_b(&Bsm[cur][(size_t)(n0w + 16 * j) * TPITCH]);

#pragma unroll
    for (int i = 0; i < 2; ++i)
#pragma unroll
      for (int j = 0; j < 4; ++j)
        c[i][j] = __builtin_amdgcn_wmma_f32_16x16x32_bf16(
            false, af[i], false, bfr[j], (short)0, c[i][j], false, false);

    __syncthreads();   // all waves done reading `cur` before it is re-staged
    buf = nxt;
  }

  // C/D layout: lane L: n=L&15, VGPR r holds row m = r + (L<16?0:8)
  const int l = lane_id();
  const int ncol  = l & 15;
  const int mhalf = (l < 16) ? 0 : 8;
#pragma unroll
  for (int i = 0; i < 2; ++i) {
#pragma unroll
    for (int j = 0; j < 4; ++j) {
#pragma unroll
      for (int r = 0; r < 8; ++r) {
        const int m = mBlk + m0w + 16 * i + mhalf + r;
        const int n = nBlk + n0w + 16 * j + ncol;
        const float v = c[i][j][r] + bias[n];
        if (mode == 0) {
          outB[(size_t)m * D_ + n] = (__bf16)v;
        } else if (mode == 1) {
          const int bb = m >> 11;          // m / S_
          const int s  = m & (S_ - 1);
          const int h  = n >> 7;           // n / DK_
          const int dk = n & (DK_ - 1);
          outB[(((size_t)(bb * H_ + h)) * DK_ + dk) * S_ + s] = (__bf16)v;
        } else {
          outF[(size_t)m * D_ + n] = v;
        }
      }
    }
  }
}

// ---------------------------------------------------------------------------
// Scores: raw = Q_bh @ K_bh^T * 1/sqrt(DK), lower-triangle tiles only.
// Causality itself is applied by the softmax kernel (reads j<=i, zeros j>i).
// Grid: (S/128 k-tiles, S/128 q-tiles, B*H). Block 8 waves (4 q x 2 k).
// DK loop is only 4 steps; direct bf16 b128 fragment loads (L2-resident).
// ---------------------------------------------------------------------------
__global__ void __launch_bounds__(256)
scores_kernel(const __bf16* __restrict__ Qb, const __bf16* __restrict__ Kb,
              float* __restrict__ attn)
{
  const int kt = blockIdx.x, qt = blockIdx.y, bh = blockIdx.z;
  if (kt > qt) return;  // uniform: whole block, EXEC stays full for WMMA
  const int bb = bh >> 4;       // bh / H_
  const int h  = bh & (H_ - 1);
  const int wave = threadIdx.x >> 5;
  const int q0  = qt * 128 + (wave >> 1) * 32;
  const int kn0 = kt * 128 + (wave & 1) * 64;

  const __bf16* Qbase = Qb + ((size_t)(bb * S_ + q0)) * D_ + h * DK_;
  const __bf16* Kbase = Kb + ((size_t)(bb * S_ + kn0)) * D_ + h * DK_;

  v8f c[2][4];
#pragma unroll
  for (int i = 0; i < 2; ++i)
#pragma unroll
    for (int j = 0; j < 4; ++j) c[i][j] = vzero8();

#pragma unroll
  for (int k0 = 0; k0 < DK_; k0 += 32) {
    v16bf af[2], bfr[4];
    af[0] = load_a_bf16(Qbase + k0, D_);
    af[1] = load_a_bf16(Qbase + (size_t)16 * D_ + k0, D_);
#pragma unroll
    for (int j = 0; j < 4; ++j)
      bfr[j] = load_b_bf16(Kbase + (size_t)(16 * j) * D_ + k0, D_);
#pragma unroll
    for (int i = 0; i < 2; ++i)
#pragma unroll
      for (int j = 0; j < 4; ++j)
        c[i][j] = __builtin_amdgcn_wmma_f32_16x16x32_bf16(
            false, af[i], false, bfr[j], (short)0, c[i][j], false, false);
  }

  const float scale = 0.08838834764831845f;  // 1/sqrt(128)
  const int l = lane_id();
  const int ncol  = l & 15;
  const int mhalf = (l < 16) ? 0 : 8;
#pragma unroll
  for (int i = 0; i < 2; ++i)
#pragma unroll
    for (int j = 0; j < 4; ++j)
#pragma unroll
      for (int r = 0; r < 8; ++r) {
        const int q = q0 + 16 * i + mhalf + r;
        const int k = kn0 + 16 * j + ncol;
        attn[((size_t)bh * S_ + q) * S_ + k] = c[i][j][r] * scale;
      }
}

// ---------------------------------------------------------------------------
// Row softmax with causal truncation. One 256-thread block per (bh, row i).
// Each thread holds 8 elements in registers (256*8 = 2048 = S).
// Reads scores only for j<=i; writes normalized prob for j<=i, 0 for j>i.
// ---------------------------------------------------------------------------
__global__ void __launch_bounds__(256)
softmax_kernel(float* __restrict__ attn)
{
  const int i  = blockIdx.x;
  const int bh = blockIdx.y;
  float* row = attn + ((size_t)bh * S_ + i) * S_;
  const int len = i + 1;
  const int t = threadIdx.x;

  float x[8];
#pragma unroll
  for (int s = 0; s < 8; ++s) {
    const int j = t + 256 * s;
    x[s] = (j < len) ? row[j] : -1.0e30f;
  }

  __shared__ float redm[8];
  __shared__ float reds[8];

  // max reduction
  float m = x[0];
#pragma unroll
  for (int s = 1; s < 8; ++s) m = fmaxf(m, x[s]);
#pragma unroll
  for (int off = 16; off >= 1; off >>= 1) m = fmaxf(m, __shfl_xor(m, off, 32));
  if ((t & 31) == 0) redm[t >> 5] = m;
  __syncthreads();
  float mx = redm[0];
#pragma unroll
  for (int w = 1; w < 8; ++w) mx = fmaxf(mx, redm[w]);

  // exp + sum reduction
  float sum = 0.0f;
#pragma unroll
  for (int s = 0; s < 8; ++s) {
    const float e = expf(x[s] - mx);   // -1e30 underflows to 0
    x[s] = e;
    sum += e;
  }
#pragma unroll
  for (int off = 16; off >= 1; off >>= 1) sum += __shfl_xor(sum, off, 32);
  if ((t & 31) == 0) reds[t >> 5] = sum;
  __syncthreads();
  float tot = 0.0f;
#pragma unroll
  for (int w = 0; w < 8; ++w) tot += reds[w];
  const float inv = 1.0f / tot;

#pragma unroll
  for (int s = 0; s < 8; ++s) {
    const int j = t + 256 * s;
    row[j] = (j < len) ? x[s] * inv : 0.0f;
  }
}

// ---------------------------------------------------------------------------
// X_bh = attn_bh (f32, converted to bf16 on load) @ V_bh, using Vt[b][h][dk][s]
// so B fragments are contiguous along k. K-loop truncated at the diagonal
// (attn is exactly 0 beyond it). Output bf16 X[b*S+q][h*DK+dk].
// Grid: (S/128, B*H). Block 8 waves (4 q x 2 n), N = DK = 128.
// ---------------------------------------------------------------------------
__global__ void __launch_bounds__(256)
attn_v_kernel(const float* __restrict__ attn, const __bf16* __restrict__ Vt,
              __bf16* __restrict__ X)
{
  const int bh = blockIdx.y;
  const int bb = bh >> 4;
  const int h  = bh & (H_ - 1);
  const int wave = threadIdx.x >> 5;
  const int q0 = blockIdx.x * 128 + (wave >> 1) * 32;
  const int n0 = (wave & 1) * 64;

  const float*  Abase = attn + ((size_t)bh * S_ + q0) * S_;
  const __bf16* Bbase = Vt + (size_t)bh * DK_ * S_ + (size_t)n0 * S_;

  v8f c[2][4];
#pragma unroll
  for (int i = 0; i < 2; ++i)
#pragma unroll
    for (int j = 0; j < 4; ++j) c[i][j] = vzero8();

  const int kend = q0 + 32;  // last needed k is q0+31 (causal)
  for (int k0 = 0; k0 < kend; k0 += 32) {
    v16bf af[2], bfr[4];
    af[0] = load_a_f32(Abase + k0, S_);
    af[1] = load_a_f32(Abase + (size_t)16 * S_ + k0, S_);
#pragma unroll
    for (int j = 0; j < 4; ++j)
      bfr[j] = load_b_bf16(Bbase + (size_t)(16 * j) * S_ + k0, S_);
#pragma unroll
    for (int i = 0; i < 2; ++i)
#pragma unroll
      for (int j = 0; j < 4; ++j)
        c[i][j] = __builtin_amdgcn_wmma_f32_16x16x32_bf16(
            false, af[i], false, bfr[j], (short)0, c[i][j], false, false);
  }

  const int l = lane_id();
  const int ncol  = l & 15;
  const int mhalf = (l < 16) ? 0 : 8;
#pragma unroll
  for (int i = 0; i < 2; ++i)
#pragma unroll
    for (int j = 0; j < 4; ++j)
#pragma unroll
      for (int r = 0; r < 8; ++r) {
        const int q  = q0 + 16 * i + mhalf + r;
        const int dk = n0 + 16 * j + ncol;
        X[((size_t)(bb * S_ + q)) * D_ + h * DK_ + dk] = (__bf16)c[i][j][r];
      }
}

// ---------------------------------------------------------------------------
extern "C" void kernel_launch(void* const* d_in, const int* in_sizes, int n_in,
                              void* d_out, int out_size, void* d_ws, size_t ws_size,
                              hipStream_t stream)
{
  (void)in_sizes; (void)n_in; (void)out_size; (void)ws_size;
  const float* query = (const float*)d_in[0];
  const float* key   = (const float*)d_in[1];
  const float* value = (const float*)d_in[2];
  const float* Wq = (const float*)d_in[3];
  const float* bq = (const float*)d_in[4];
  const float* Wk = (const float*)d_in[5];
  const float* bk = (const float*)d_in[6];
  const float* Wv = (const float*)d_in[7];
  const float* bv = (const float*)d_in[8];
  const float* Wo = (const float*)d_in[9];
  const float* bo = (const float*)d_in[10];

  float* out  = (float*)d_out;
  float* attn = out + (size_t)B_ * S_ * D_;   // [B,H,S,S]

  const size_t elems = (size_t)B_ * S_ * D_;  // 8.39M
  __bf16* Qb = (__bf16*)d_ws;                 // [B*S, D] bf16
  __bf16* Kb = Qb + elems;                    // [B*S, D] bf16
  __bf16* Vt = Kb + elems;                    // [B,H,DK,S] bf16
  __bf16* Xb = Vt + elems;                    // [B*S, D] bf16   (total 64 MB)

  dim3 blk(256);
  dim3 gproj(D_ / 128, (B_ * S_) / 128);      // (16, 32)

  proj_gemm_kernel<<<gproj, blk, 0, stream>>>(query, nullptr, Wq, bq, Qb, nullptr, 0);
  proj_gemm_kernel<<<gproj, blk, 0, stream>>>(key,   nullptr, Wk, bk, Kb, nullptr, 0);
  proj_gemm_kernel<<<gproj, blk, 0, stream>>>(value, nullptr, Wv, bv, Vt, nullptr, 1);

  dim3 gsc(S_ / 128, S_ / 128, B_ * H_);      // (16,16,32), upper tiles early-exit
  scores_kernel<<<gsc, blk, 0, stream>>>(Qb, Kb, attn);

  dim3 gsm(S_, B_ * H_);                      // (2048, 32)
  softmax_kernel<<<gsm, blk, 0, stream>>>(attn);

  dim3 gav(S_ / 128, B_ * H_);                // (16, 32)
  attn_v_kernel<<<gav, blk, 0, stream>>>(attn, Vt, Xb);

  proj_gemm_kernel<<<gproj, blk, 0, stream>>>(nullptr, Xb, Wo, bo, nullptr, out, 2);
}
